// Model_53618371723669
// MI455X (gfx1250) — matrix operand order repaired and checked by reading, compiled, not run on hardware
//
#include <hip/hip_runtime.h>

// ---------------------------------------------------------------------------
// Persistent graph-LSTM kernel for MI455X (gfx1250, wave32, WMMA).
// One block = 8 batch items, entire 88-step recurrence in LDS.
// Recurrent matmuls AND the decode projection both run on v_wmma.
// ---------------------------------------------------------------------------

typedef _Float16 v16h __attribute__((ext_vector_type(16)));
typedef _Float16 v8h  __attribute__((ext_vector_type(8)));
typedef _Float16 v2h  __attribute__((ext_vector_type(2)));
typedef float    v8f  __attribute__((ext_vector_type(8)));

#define JN   24          // joints
#define HD   24          // hidden
#define BI   8           // batch items per block
#define ROWS (BI*JN)     // 192 node-rows per block
#define MTILES (ROWS/16) // 12
#define AKS  136         // Acomb row stride in halves (272B, bank-skewed)
#define WKS  136         // weight K stride in halves
#define PKS  584         // projection row stride in halves (1168B, bank-skewed)
#define TSRC 32
#define TOUT 64
#define WARM 24
#define NTHREADS 256

// SMPL tree neighborhoods (self first, then parent + children), padded to 5
__constant__ int c_nbr[JN][5] = {
  {0,1,2,3,0},{1,0,4,0,0},{2,0,5,0,0},{3,0,6,0,0},{4,1,7,0,0},{5,2,8,0,0},
  {6,3,9,0,0},{7,4,10,0,0},{8,5,11,0,0},{9,6,12,13,14},{10,7,0,0,0},{11,8,0,0,0},
  {12,9,15,0,0},{13,9,16,0,0},{14,9,17,0,0},{15,12,0,0,0},{16,13,18,0,0},
  {17,14,19,0,0},{18,16,20,0,0},{19,17,21,0,0},{20,18,22,0,0},{21,19,23,0,0},
  {22,20,0,0,0},{23,21,0,0,0}
};
__constant__ int c_ncnt[JN] = {4,3,3,3,3,3,3,3,3,5,2,2,3,3,3,2,3,3,3,3,3,3,2,2};
__constant__ float c_invdeg[JN] = {
  0.25f, 1.f/3.f, 1.f/3.f, 1.f/3.f, 1.f/3.f, 1.f/3.f, 1.f/3.f, 1.f/3.f, 1.f/3.f,
  0.2f, 0.5f, 0.5f, 1.f/3.f, 1.f/3.f, 1.f/3.f, 0.5f, 1.f/3.f, 1.f/3.f, 1.f/3.f,
  1.f/3.f, 1.f/3.f, 1.f/3.f, 0.5f, 0.5f
};

// LDS byte layout (all offsets 16B aligned)
#define OFF_ACOMB 0
#define SZ_ACOMB  (ROWS*AKS*2)                 // 52224
#define OFF_WC    (OFF_ACOMB + SZ_ACOMB)
#define SZ_WC     (4*32*WKS*2)                 // 34816
#define OFF_OUT   (OFF_WC + SZ_WC)
#define SZ_OUT    (ROWS*HD*4)                  // 18432
#define OFF_H     (OFF_OUT + SZ_OUT)
#define SZ_H      (ROWS*HD*4)
#define OFF_C     (OFF_H + SZ_H)
#define SZ_C      (ROWS*HD*4)
#define OFF_BIAS  (OFF_C + SZ_C)
#define SZ_BIAS   (4*HD*4)
#define OFF_PEEP  (OFF_BIAS + SZ_BIAS)
#define SZ_PEEP   (3*HD*4 + 96)
#define OFF_PW    (OFF_PEEP + SZ_PEEP)
#define SZ_PW     (16*PKS*2)                   // 18688 (8 items x 2 steps staging)
#define OFF_DW    (OFF_PW + SZ_PW)
#define SZ_DW     (80*PKS*2)                   // 93440 (dec_W in f16, 72->80 pad)
#define SMEM_BYTES (OFF_DW + SZ_DW)            // 255232 B  (< 320KB/WGP)

#if __has_builtin(__builtin_amdgcn_rcpf)
#define RCPF(x) __builtin_amdgcn_rcpf(x)
#else
#define RCPF(x) (1.0f / (x))
#endif

__device__ __forceinline__ float sigmoidf_(float x) {
  return RCPF(1.0f + __expf(-x));      // 1 v_exp + 1 v_rcp
}
__device__ __forceinline__ float tanhf_(float x) {
  float e = __expf(2.0f * x);          // 1 v_exp + 1 v_rcp + 2 valu
  return (e - 1.0f) * RCPF(e + 1.0f);
}

// Build a 16-half fragment from two 8-half LDS chunks (two ds_load_b128)
__device__ __forceinline__ v16h ld2x8(const _Float16* p0, const _Float16* p1) {
  v8h a = *(const v8h*)p0;
  v8h b = *(const v8h*)p1;
  return __builtin_shufflevector(a, b, 0,1,2,3,4,5,6,7,8,9,10,11,12,13,14,15);
}

// ---- phase helpers (called under uniform control flow, barriers outside) ----

__device__ __forceinline__ void encode_phase(float* outb, const float* x,
                                             long itemStride, const float* encW,
                                             const float* encb, int n0, int tid) {
  for (int idx = tid; idx < ROWS*HD; idx += NTHREADS) {
    int row = idx / HD, ch = idx % HD;
    int i = row / JN, j = row % JN;
    const float* xp = x + (long)(n0 + i) * itemStride + j*3;
    float v = encb[ch] + xp[0]*encW[ch*3+0] + xp[1]*encW[ch*3+1] + xp[2]*encW[ch*3+2];
    outb[row*HD + ch] = v > 0.0f ? v : 0.0f;
  }
}

// Vectorized (channel-pair) pack: builds all four f16 column blocks of Acomb.
// Thread -> (row stripe, channel pair); indices carried incrementally (no div).
__device__ __forceinline__ void pack_phase(_Float16* Acomb, const float* outb,
                                           const float* hbuf, int tid) {
  const int p = tid & 15;          // channel pair 0..15 (12 valid)
  if (p < 12) {
    int row = tid >> 4;            // 0..15, then +16 each iteration
    int j   = row;                 // row < 24 here, so j == row
    for (int k = 0; k < 12; ++k) {
      const int basei = row - j;   // item_index * JN
      const float* ob = outb + basei*HD + 2*p;
      const float* hb = hbuf + basei*HD + 2*p;
      // self neighbor first (c_nbr[j][0] == j): doubles as X/h column data
      const float2 xs = *(const float2*)(ob + j*HD);
      const float2 hsv = *(const float2*)(hb + j*HD);
      float ax0 = xs.x, ax1 = xs.y, ah0 = hsv.x, ah1 = hsv.y;
      const int cnt = c_ncnt[j];
      for (int t = 1; t < cnt; ++t) {
        const int nb = c_nbr[j][t] * HD;
        const float2 xv = *(const float2*)(ob + nb);
        const float2 hv = *(const float2*)(hb + nb);
        ax0 += xv.x; ax1 += xv.y; ah0 += hv.x; ah1 += hv.y;
      }
      const float inv = c_invdeg[j];
      _Float16* ar = Acomb + row*AKS + 2*p;
      *(v2h*)(ar +  0) = (v2h){(_Float16)(ax0*inv), (_Float16)(ax1*inv)};
      *(v2h*)(ar + 32) = (v2h){(_Float16)xs.x,      (_Float16)xs.y};
      *(v2h*)(ar + 64) = (v2h){(_Float16)(ah0*inv), (_Float16)(ah1*inv)};
      *(v2h*)(ar + 96) = (v2h){(_Float16)hsv.x,     (_Float16)hsv.y};
      row += 16;
      j += 16; if (j >= JN) j -= JN;
    }
  }
}

// Recurrent step: 4-gate fused WMMA [192,128]x[128,32] + LSTM gate math in
// registers. Balanced: 24 (M-tile,N-tile) units over 8 waves = 3 per wave.
// Unit loop kept ROLLED so the scheduler can deep-pipeline one compact body.
template<bool STAGE>
__device__ __forceinline__ void step_phase(const _Float16* Acomb, const _Float16* Wc,
                                           const float* biasG, const float* peep,
                                           float* outb, float* hbuf, float* cbuf,
                                           _Float16* Pw, int pslot, int tid) {
  const int wave = tid >> 5;
  const int lane = tid & 31;
  const int hs   = lane >> 4;   // half-wave select
  const int l15  = lane & 15;

  #pragma unroll 1
  for (int ui = 0; ui < 3; ++ui) {
    const int u  = wave*3 + ui;
    const int mt = u >> 1;
    const int nt = u & 1;
    const int mb = mt * 16;
    // A fragments: ISA layout — lanes 0-15: K {0..7,16..23}; lanes 16-31: K {8..15,24..31}
    v16h af[4];
    #pragma unroll
    for (int k = 0; k < 4; ++k) {
      const _Float16* pa = Acomb + (mb + l15)*AKS + k*32 + hs*8;
      af[k] = ld2x8(pa, pa + 16);
    }
    v8f acc[4];
    #pragma unroll
    for (int g = 0; g < 4; ++g) {
      v8f a = {};
      #pragma unroll
      for (int k = 0; k < 4; ++k) {
        // B fragment: lane = output column, K {0..7,16..23} / {8..15,24..31} per half-wave
        const _Float16* bp = Wc + (g*32 + nt*16 + l15)*WKS + k*32 + hs*8;
        v16h bf = ld2x8(bp, bp + 16);
        a = __builtin_amdgcn_wmma_f32_16x16x32_f16(
              false, af[k], false, bf, (short)0, a, false, false);
      }
      acc[g] = a;
    }
    // LSTM gate math directly on accumulators (D layout: reg r, lane -> M,N)
    const int col = nt*16 + l15;
    if (col < HD) {
      const float bI = biasG[0*HD+col], bF = biasG[1*HD+col];
      const float bC = biasG[2*HD+col], bO = biasG[3*HD+col];
      const float pI = peep[0*HD+col],  pF = peep[1*HD+col], pO = peep[2*HD+col];
      #pragma unroll
      for (int r = 0; r < 8; ++r) {
        const int row = mb + r + hs*8;
        const float cp = cbuf[row*HD + col];
        const float ig = sigmoidf_(acc[0][r] + pI*cp + bI);
        const float fg = sigmoidf_(acc[1][r] + pF*cp + bF);
        const float c2 = fg*cp + ig*tanhf_(acc[2][r] + bC);
        const float og = sigmoidf_(acc[3][r] + pO*c2 + bO);
        const float h2 = og * tanhf_(c2);
        cbuf[row*HD + col] = c2;
        hbuf[row*HD + col] = h2;
        outb[row*HD + col] = og;   // X_next = output gate value
        if (STAGE) {
          int i = row / JN, j = row - i*JN;
          Pw[(pslot*BI + i)*PKS + j*HD + col] = (_Float16)og;
        }
      }
    }
  }
}

// WMMA projection of two staged decode steps: [16,576] x [576,80].
// Waves 0..4 each own one 16-col N-tile (18 K-steps of 32).
__device__ __forceinline__ void project_wmma(const _Float16* Pw, const _Float16* dWf,
                                             const float* decb, float* out,
                                             int n0, int tbase, int tid) {
  const int wave = tid >> 5;
  const int lane = tid & 31;
  const int hs   = lane >> 4;
  const int l15  = lane & 15;
  if (wave < 5) {
    const int nt = wave;
    v8f acc = {};
    #pragma unroll 6
    for (int k = 0; k < 18; ++k) {
      const _Float16* ap = Pw + l15*PKS + k*32 + hs*8;
      v16h af = ld2x8(ap, ap + 16);
      const _Float16* bp = dWf + (nt*16 + l15)*PKS + k*32 + hs*8;
      v16h bf = ld2x8(bp, bp + 16);
      acc = __builtin_amdgcn_wmma_f32_16x16x32_f16(
              false, af, false, bf, (short)0, acc, false, false);
    }
    const int col = nt*16 + l15;
    if (col < 72) {
      const float bb = decb[col];
      #pragma unroll
      for (int r = 0; r < 8; ++r) {
        const int M = r + hs*8;         // staged row: (step parity)*8 + item
        const int item = M & 7;
        const int tp = tbase + (M >> 3);
        out[(((long)(n0 + item))*TOUT + tp)*72 + col] = acc[r] + bb;
      }
    }
  }
}

__global__ void __launch_bounds__(NTHREADS)
graph_lstm_kernel(const float* __restrict__ src, const float* __restrict__ tgt,
                  const float* __restrict__ encW, const float* __restrict__ encb,
                  const float* __restrict__ sxWl, const float* __restrict__ sxWr,
                  const float* __restrict__ sxb,  const float* __restrict__ shWl,
                  const float* __restrict__ shWr, const float* __restrict__ shb,
                  const float* __restrict__ gatew, const float* __restrict__ gateb,
                  const float* __restrict__ decW, const float* __restrict__ decb,
                  float* __restrict__ out) {
  extern __shared__ char smem[];
  _Float16* Acomb = (_Float16*)(smem + OFF_ACOMB);
  _Float16* Wc    = (_Float16*)(smem + OFF_WC);
  float*    outb  = (float*)(smem + OFF_OUT);
  float*    hbuf  = (float*)(smem + OFF_H);
  float*    cbuf  = (float*)(smem + OFF_C);
  float*    biasG = (float*)(smem + OFF_BIAS);
  float*    peep  = (float*)(smem + OFF_PEEP);
  _Float16* Pw    = (_Float16*)(smem + OFF_PW);
  _Float16* dWf   = (_Float16*)(smem + OFF_DW);

  const int tid = threadIdx.x;
  const int n0  = blockIdx.x * BI;

  // ---- init: zero Acomb/Pw padding, pack weights f32->f16, fold biases ----
  for (int idx = tid; idx < ROWS*AKS; idx += NTHREADS) Acomb[idx] = (_Float16)0.0f;
  for (int idx = tid; idx < 16*PKS; idx += NTHREADS) Pw[idx] = (_Float16)0.0f;
  for (int idx = tid; idx < 4*32*WKS; idx += NTHREADS) {
    int g = idx / (32*WKS), rem = idx % (32*WKS);
    int n = rem / WKS, k = rem % WKS;
    float v = 0.0f;
    if (k < 128 && n < HD) {
      int kb = k >> 5, h = k & 31;
      if (h < HD) {
        int o = g*576 + n*24 + h;
        v = (kb == 0) ? sxWl[o] : (kb == 1) ? sxWr[o] : (kb == 2) ? shWl[o] : shWr[o];
      }
    }
    Wc[(g*32 + n)*WKS + k] = (_Float16)v;
  }
  for (int idx = tid; idx < 80*PKS; idx += NTHREADS) {
    int n = idx / PKS, k = idx % PKS;
    float v = (n < 72 && k < 576) ? decW[n*576 + k] : 0.0f;
    dWf[idx] = (_Float16)v;
  }
  for (int idx = tid; idx < 4*HD; idx += NTHREADS) {
    int g = idx / HD, c = idx % HD;
    biasG[idx] = sxb[g*HD+c] + shb[g*HD+c] + gateb[g*HD+c];
  }
  for (int idx = tid; idx < 3*HD; idx += NTHREADS) peep[idx] = gatew[idx];
  for (int idx = tid; idx < ROWS*HD; idx += NTHREADS) { cbuf[idx] = 0.0f; hbuf[idx] = 0.0f; }
  // X0 = enc(src[:, T_src-WARM]) == src frame 8
  encode_phase(outb, src + (long)(TSRC - WARM)*JN*3, (long)TSRC*JN*3, encW, encb, n0, tid);
  __syncthreads();

  // ---- warmup: 24 recurrent steps (no projection, no staging) ----
  for (int s = 0; s < WARM; ++s) {
    pack_phase(Acomb, outb, hbuf, tid);
    __syncthreads();
    step_phase<false>(Acomb, Wc, biasG, peep, outb, hbuf, cbuf, Pw, 0, tid);
    __syncthreads();
  }

  // ---- re-encode: X = enc(tgt[:, 0]) (h/c carry over from warmup) ----
  encode_phase(outb, tgt, (long)TOUT*JN*3, encW, encb, n0, tid);
  __syncthreads();

  // ---- autoregressive decode: 64 steps; WMMA projection every 2 steps ----
  for (int t = 0; t < TOUT; ++t) {
    pack_phase(Acomb, outb, hbuf, tid);
    __syncthreads();
    step_phase<true>(Acomb, Wc, biasG, peep, outb, hbuf, cbuf, Pw, t & 1, tid);
    __syncthreads();
    if (t & 1) {
      project_wmma(Pw, dWf, decb, out, n0, t - 1, tid);
      __syncthreads();
    }
  }
}

extern "C" void kernel_launch(void* const* d_in, const int* in_sizes, int n_in,
                              void* d_out, int out_size, void* d_ws, size_t ws_size,
                              hipStream_t stream) {
  (void)in_sizes; (void)n_in; (void)out_size; (void)d_ws; (void)ws_size;
  const float* src   = (const float*)d_in[0];
  const float* tgt   = (const float*)d_in[1];
  const float* encW  = (const float*)d_in[2];
  const float* encb  = (const float*)d_in[3];
  const float* sxWl  = (const float*)d_in[4];
  const float* sxWr  = (const float*)d_in[5];
  const float* sxb   = (const float*)d_in[6];
  const float* shWl  = (const float*)d_in[7];
  const float* shWr  = (const float*)d_in[8];
  const float* shb   = (const float*)d_in[9];
  const float* gatew = (const float*)d_in[10];
  const float* gateb = (const float*)d_in[11];
  const float* decW  = (const float*)d_in[12];
  const float* decb  = (const float*)d_in[13];
  float* out = (float*)d_out;

  (void)hipFuncSetAttribute((const void*)graph_lstm_kernel,
                            hipFuncAttributeMaxDynamicSharedMemorySize, SMEM_BYTES);

  dim3 grid(1024 / BI);   // 128 persistent blocks x 8 items
  dim3 block(NTHREADS);   // 8 wave32 waves
  graph_lstm_kernel<<<grid, block, SMEM_BYTES, stream>>>(
      src, tgt, encW, encb, sxWl, sxWr, sxb, shWl, shWr, shb,
      gatew, gateb, decW, decb, out);
}
